// Transformer1_39797166965112
// MI455X (gfx1250) — compile-verified, matrix operand order checked by reading, not run on hardware
//
#include <hip/hip_runtime.h>
#include <hip/hip_bf16.h>
#include <math.h>

// ---------------------------------------------------------------------------
// Types for CDNA5 WMMA
// ---------------------------------------------------------------------------
typedef __attribute__((ext_vector_type(16))) __bf16 bf16x16;
typedef __attribute__((ext_vector_type(8)))  float  f32x8;

union Frag { uint4 q[2]; bf16x16 v; };

__device__ __forceinline__ unsigned f2bf_bits(float f) {
  union { float f; unsigned u; } in; in.f = f;
  return (in.u + 0x7FFFu + ((in.u >> 16) & 1u)) >> 16;   // RNE
}
__device__ __forceinline__ unsigned pack2bf(float lo, float hi) {
  return f2bf_bits(lo) | (f2bf_bits(hi) << 16);
}

// ---------------------------------------------------------------------------
// Generic batched GEMM:  C[z] = alpha * A[z] (x) B[z]  (+bias) (relu) (+Cadd)
// A: MxK row-major (lda), B: KxN row-major (ldb)  [TRANSB: B is NxK, B^T used]
// Batch z = z1*Z2 + z2; per-pointer offsets z1*Pz1 + z2*Pz2 (Cadd shares C's).
// 128 threads = 4 waves; block tile 64x64; wave tile 32x32 (4 x 16x16 WMMA,
// 2x A-frag and 2x B-frag reuse). Tiles staged in LDS K-innermost so each
// lane's fragment is two contiguous 16B chunks -> two ds_load_b128.
// Requires K % 32 == 0, lda/ldb(TRANSB) % 4 == 0 (true for all call sites).
// ---------------------------------------------------------------------------
#define ROWU 20   // LDS row stride in uints: 32 bf16 (16u) + 4u pad = 80B (16B-aligned)

template<bool TRANSB>
__global__ __launch_bounds__(128)
void gemm_bf16_wmma(const float* __restrict__ A, const float* __restrict__ Bm,
                    const float* __restrict__ bias, const float* __restrict__ Cadd,
                    float* __restrict__ C,
                    int M, int N, int K,
                    long lda, long ldb, long ldc,
                    int Z2,
                    long Az1, long Az2, long Bz1, long Bz2, long Cz1, long Cz2,
                    float alpha, int relu)
{
  __shared__ unsigned As_u[64 * ROWU];   // A tile 64(M) x 32(K), K-innermost
  __shared__ unsigned Bs_u[64 * ROWU];   // B tile 64(N) x 32(K), K-innermost

  const int z  = blockIdx.z;
  const int z1 = z / Z2, z2 = z % Z2;
  A  += (long)z1 * Az1 + (long)z2 * Az2;
  Bm += (long)z1 * Bz1 + (long)z2 * Bz2;
  C  += (long)z1 * Cz1 + (long)z2 * Cz2;
  if (Cadd) Cadd += (long)z1 * Cz1 + (long)z2 * Cz2;

  const int tile_m = blockIdx.y * 64;
  const int tile_n = blockIdx.x * 64;
  const int tid  = threadIdx.x;
  const int wid  = tid >> 5;
  const int lane = tid & 31;
  const int wr   = wid & 1;            // M half of block tile (32 rows)
  const int wc   = wid >> 1;           // N half of block tile (32 cols)
  const int lrow = lane & 15;
  const int h4   = (lane < 16) ? 0 : 1;  // uint4 chunk selector (ISA K-half)

  f32x8 acc[2][2];
  #pragma unroll
  for (int i = 0; i < 2; ++i)
    #pragma unroll
    for (int j = 0; j < 2; ++j)
      #pragma unroll
      for (int e = 0; e < 8; ++e) acc[i][j][e] = 0.f;

  for (int k0 = 0; k0 < K; k0 += 32) {
    if (k0 + 32 < K) {  // global_prefetch_b8 next K tile
      __builtin_prefetch(&A[(long)(tile_m + (tid & 63)) * lda + (k0 + 32)], 0, 1);
      if (TRANSB) __builtin_prefetch(&Bm[(long)(tile_n + (tid & 63)) * ldb + (k0 + 32)], 0, 1);
      else        __builtin_prefetch(&Bm[(long)(k0 + 32 + (tid & 31)) * ldb + tile_n], 0, 1);
    }

    // ---- stage A: 64 rows x 8 groups of 4 K; float4 load -> packed bf16 b64 store
    #pragma unroll
    for (int it = 0; it < 4; ++it) {
      int item = it * 128 + tid;
      int r = item >> 3, kg = item & 7;
      int gm  = tile_m + r;
      int gmc = (gm < M) ? gm : (M - 1);
      float4 v = *(const float4*)(A + (long)gmc * lda + (k0 + kg * 4));
      if (gm >= M) { v.x = 0.f; v.y = 0.f; v.z = 0.f; v.w = 0.f; }
      *(uint2*)(As_u + r * ROWU + kg * 2) = make_uint2(pack2bf(v.x, v.y), pack2bf(v.z, v.w));
    }

    // ---- stage B as Bs[n][k]
    if (TRANSB) {          // B is NxK row-major: same-major as A staging
      #pragma unroll
      for (int it = 0; it < 4; ++it) {
        int item = it * 128 + tid;
        int r = item >> 3, kg = item & 7;
        int gn  = tile_n + r;
        int gnc = (gn < N) ? gn : (N - 1);
        float4 v = *(const float4*)(Bm + (long)gnc * ldb + (k0 + kg * 4));
        if (gn >= N) { v.x = 0.f; v.y = 0.f; v.z = 0.f; v.w = 0.f; }
        *(uint2*)(Bs_u + r * ROWU + kg * 2) = make_uint2(pack2bf(v.x, v.y), pack2bf(v.z, v.w));
      }
    } else {               // B is KxN row-major: 4 strided K per fixed n (coalesced over n)
      #pragma unroll
      for (int it = 0; it < 4; ++it) {
        int item = it * 128 + tid;
        int n = item & 63, kg = item >> 6;
        int gn  = tile_n + n;
        int gnc = (gn < N) ? gn : (N - 1);
        const float* bp = Bm + (long)(k0 + kg * 4) * ldb + gnc;
        float v0 = bp[0], v1 = bp[ldb], v2 = bp[2 * ldb], v3 = bp[3 * ldb];
        if (gn >= N) { v0 = 0.f; v1 = 0.f; v2 = 0.f; v3 = 0.f; }
        *(uint2*)(Bs_u + n * ROWU + kg * 2) = make_uint2(pack2bf(v0, v1), pack2bf(v2, v3));
      }
    }
    __syncthreads();

    // ---- fragments: two ds_load_b128 each (lane chunks contiguous per ISA layout)
    Frag a[2], b[2];
    #pragma unroll
    for (int i = 0; i < 2; ++i) {
      const uint4* ap = (const uint4*)(As_u + (wr * 32 + i * 16 + lrow) * ROWU);
      a[i].q[0] = ap[h4];
      a[i].q[1] = ap[h4 + 2];
      const uint4* bp = (const uint4*)(Bs_u + (wc * 32 + i * 16 + lrow) * ROWU);
      b[i].q[0] = bp[h4];
      b[i].q[1] = bp[h4 + 2];
    }
    #pragma unroll
    for (int i = 0; i < 2; ++i)
      #pragma unroll
      for (int j = 0; j < 2; ++j)
        acc[i][j] = __builtin_amdgcn_wmma_f32_16x16x32_bf16(
            false, a[i].v, false, b[j].v, (short)0, acc[i][j], false, false);
    __syncthreads();
  }

  // ---- epilogue: C 16x16 tile layout: VGPR r -> M=r (+8 lanes 16..31), N=lane&15
  const int rbase = (lane < 16) ? 0 : 8;
  const int cbase = lane & 15;
  #pragma unroll
  for (int i = 0; i < 2; ++i) {
    #pragma unroll
    for (int j = 0; j < 2; ++j) {
      const int gn = tile_n + wc * 32 + j * 16 + cbase;
      if (gn >= N) continue;
      const float bv = bias ? bias[gn] : 0.f;
      #pragma unroll
      for (int r = 0; r < 8; ++r) {
        int gm = tile_m + wr * 32 + i * 16 + rbase + r;
        if (gm >= M) continue;
        float v = acc[i][j][r] * alpha + bv;
        if (relu) v = fmaxf(v, 0.f);
        if (Cadd) v += Cadd[(long)gm * ldc + gn];
        C[(long)gm * ldc + gn] = v;
      }
    }
  }
}

// ---------------------------------------------------------------------------
// LayerNorm over last dim D (one 256-thread block per row)
// ---------------------------------------------------------------------------
__global__ __launch_bounds__(256)
void layernorm_kernel(const float* __restrict__ x, const float* __restrict__ g,
                      const float* __restrict__ b, float* __restrict__ y, int D)
{
  __shared__ float red[256];
  const long row = blockIdx.x;
  const float* xr = x + row * (long)D;
  float s = 0.f;
  for (int i = threadIdx.x; i < D; i += 256) s += xr[i];
  red[threadIdx.x] = s; __syncthreads();
  for (int off = 128; off; off >>= 1) {
    if (threadIdx.x < off) red[threadIdx.x] += red[threadIdx.x + off];
    __syncthreads();
  }
  const float mean = red[0] / (float)D;
  __syncthreads();
  float v = 0.f;
  for (int i = threadIdx.x; i < D; i += 256) { float d = xr[i] - mean; v += d * d; }
  red[threadIdx.x] = v; __syncthreads();
  for (int off = 128; off; off >>= 1) {
    if (threadIdx.x < off) red[threadIdx.x] += red[threadIdx.x + off];
    __syncthreads();
  }
  const float rstd = rsqrtf(red[0] / (float)D + 1e-5f);
  for (int i = threadIdx.x; i < D; i += 256)
    y[row * (long)D + i] = (xr[i] - mean) * rstd * g[i] + b[i];
}

// ---------------------------------------------------------------------------
// In-place softmax over last dim S (one block per row)
// ---------------------------------------------------------------------------
__global__ __launch_bounds__(256)
void softmax_kernel(float* __restrict__ x, int S)
{
  __shared__ float red[256];
  float* p = x + (long)blockIdx.x * S;
  float m = -3.0e38f;
  for (int i = threadIdx.x; i < S; i += 256) m = fmaxf(m, p[i]);
  red[threadIdx.x] = m; __syncthreads();
  for (int off = 128; off; off >>= 1) {
    if (threadIdx.x < off) red[threadIdx.x] = fmaxf(red[threadIdx.x], red[threadIdx.x + off]);
    __syncthreads();
  }
  m = red[0]; __syncthreads();
  float s = 0.f;
  for (int i = threadIdx.x; i < S; i += 256) { float e = __expf(p[i] - m); p[i] = e; s += e; }
  red[threadIdx.x] = s; __syncthreads();
  for (int off = 128; off; off >>= 1) {
    if (threadIdx.x < off) red[threadIdx.x] += red[threadIdx.x + off];
    __syncthreads();
  }
  const float inv = 1.0f / red[0];
  for (int i = threadIdx.x; i < S; i += 256) p[i] *= inv;
}

// ---------------------------------------------------------------------------
// y += x (residual add)
// ---------------------------------------------------------------------------
__global__ __launch_bounds__(256)
void add_kernel(float* __restrict__ y, const float* __restrict__ x, long n)
{
  long i = (long)blockIdx.x * 256 + threadIdx.x;
  const long stride = (long)gridDim.x * 256;
  for (; i < n; i += stride) y[i] += x[i];
}

// ---------------------------------------------------------------------------
// Encoder embedding: x[b,t,:] = emb[idx[b,t],:] + pe(t, E)
// ---------------------------------------------------------------------------
__global__ __launch_bounds__(256)
void embed_src_kernel(const float* __restrict__ emb, const int* __restrict__ idx,
                      float* __restrict__ x, int T, int E)
{
  const int t = blockIdx.x, b = blockIdx.y;
  const int token = idx[b * T + t];
  const float c0 = -logf(10000.f) / (float)E;
  for (int e = threadIdx.x; e < E; e += 256) {
    int i2 = e >> 1;
    float div = __expf((float)(2 * i2) * c0);
    float ang = (float)t * div;
    float pe = (e & 1) ? __cosf(ang) : __sinf(ang);
    x[((long)b * T + t) * E + e] = emb[(long)token * E + e] + pe;
  }
}

// ---------------------------------------------------------------------------
// Decoder embedding: concat(tok_emb+pe (E3), field_emb (E3), type_emb (E3))
// ---------------------------------------------------------------------------
__global__ __launch_bounds__(256)
void embed_dec_kernel(const float* __restrict__ tok_emb, const float* __restrict__ field_emb,
                      const float* __restrict__ type_emb,
                      const int* __restrict__ tgt, const int* __restrict__ fld,
                      const int* __restrict__ typ,
                      float* __restrict__ d, int T, int E3)
{
  const int t = blockIdx.x, b = blockIdx.y;
  const int E = 3 * E3;
  const int tk = tgt[b * T + t], fk = fld[b * T + t], yk = typ[b * T + t];
  float* row = d + ((long)b * T + t) * E;
  const float c0 = -logf(10000.f) / (float)E3;
  for (int e = threadIdx.x; e < E3; e += 256) {
    int i2 = e >> 1;
    float div = __expf((float)(2 * i2) * c0);
    float ang = (float)t * div;
    float pe = (e & 1) ? __cosf(ang) : __sinf(ang);
    row[e]          = tok_emb[(long)tk * E3 + e] + pe;
    row[E3 + e]     = field_emb[(long)fk * E3 + e];
    row[2 * E3 + e] = type_emb[(long)yk * E3 + e];
  }
}

// ---------------------------------------------------------------------------
// gen_copy head: log_softmax(d @ W(E,2) + b) -> out (rows, 2)
// ---------------------------------------------------------------------------
__global__ __launch_bounds__(256)
void gen_copy_kernel(const float* __restrict__ d, const float* __restrict__ W,
                     const float* __restrict__ bias, float* __restrict__ out, int E)
{
  __shared__ float s0[256], s1[256];
  const long row = blockIdx.x;
  const float* x = d + row * (long)E;
  float a0 = 0.f, a1 = 0.f;
  for (int i = threadIdx.x; i < E; i += 256) {
    float xv = x[i];
    a0 += xv * W[i * 2 + 0];
    a1 += xv * W[i * 2 + 1];
  }
  s0[threadIdx.x] = a0; s1[threadIdx.x] = a1; __syncthreads();
  for (int off = 128; off; off >>= 1) {
    if (threadIdx.x < off) { s0[threadIdx.x] += s0[threadIdx.x + off];
                             s1[threadIdx.x] += s1[threadIdx.x + off]; }
    __syncthreads();
  }
  if (threadIdx.x == 0) {
    float v0 = s0[0] + bias[0], v1 = s1[0] + bias[1];
    float m = fmaxf(v0, v1);
    float lse = m + logf(__expf(v0 - m) + __expf(v1 - m));
    out[row * 2 + 0] = v0 - lse;
    out[row * 2 + 1] = v1 - lse;
  }
}

// ---------------------------------------------------------------------------
// Bahdanau pointer: out[b,tq,tk] = sum_j tanh(qp[b,tq,j]+kp[b,tk,j])*w2[j] + b2
// ---------------------------------------------------------------------------
__global__ __launch_bounds__(256)
void ptr_kernel(const float* __restrict__ qp, const float* __restrict__ kp,
                const float* __restrict__ w2, const float* __restrict__ b2,
                float* __restrict__ out, int Tq, int Tk, int P)
{
  __shared__ float qs[64], ws[64];
  const int tq = blockIdx.x, b = blockIdx.y;
  if ((int)threadIdx.x < P) {
    qs[threadIdx.x] = qp[((long)b * Tq + tq) * P + threadIdx.x];
    ws[threadIdx.x] = w2[threadIdx.x];
  }
  __syncthreads();
  const float bb = b2[0];
  for (int tk = threadIdx.x; tk < Tk; tk += 256) {
    const float* kr = kp + ((long)b * Tk + tk) * P;
    float s = bb;
    #pragma unroll 10
    for (int j = 0; j < P; ++j) s += tanhf(qs[j] + kr[j]) * ws[j];
    out[((long)b * Tq + tq) * Tk + tk] = s;
  }
}

// ---------------------------------------------------------------------------
// Host orchestration
// ---------------------------------------------------------------------------
extern "C" void kernel_launch(void* const* d_in, const int* in_sizes, int n_in,
                              void* d_out, int out_size, void* d_ws, size_t ws_size,
                              hipStream_t stream)
{
  (void)in_sizes; (void)n_in; (void)out_size; (void)ws_size;

  const int Bn = 8, T = 512, E = 768, Hh = 4, HD = 192, FFd = 3072;
  const int E3 = 256, PROJ = 50, TGTV = 512, PRIMV = 10000;
  const int M = Bn * T;  // 4096

  // Leaf mapping (JAX pytree order: sorted dict keys, 'W' < 'b'):
  // dec[l] (16): 0 ca.Wk 1 ca.Wq 2 ca.Wv 3 ff.W1 4 ff.W2 5 ff.b1 6 ff.b2
  //              7 ln1.b 8 ln1.g 9 ln2.b 10 ln2.g 11 ln3.b 12 ln3.g
  //              13 sa.Wk 14 sa.Wq 15 sa.Wv
  // enc[l] (11): 0 ff.W1 1 ff.W2 2 ff.b1 3 ff.b2 4 ln1.b 5 ln1.g
  //              6 ln2.b 7 ln2.g 8 sa.Wk 9 sa.Wq 10 sa.Wv
  auto Fp = [&](int i) { return (const float*)d_in[i]; };
  const int* src_idx   = (const int*)d_in[124];
  const int* tgt_idx   = (const int*)d_in[125];
  const int* field_idx = (const int*)d_in[126];
  const int* type_idx  = (const int*)d_in[127];

  float* ws = (float*)d_ws;
  size_t off = 0;
  auto alloc = [&](size_t n) { float* p = ws + off; off += n; return p; };
  const size_t SME = (size_t)M * E;
  float* xe      = alloc(SME);
  float* lnb     = alloc(SME);
  float* lnkv    = alloc(SME);
  float* qb      = alloc(SME);              // (B,H,T,HD)
  float* kb      = alloc(SME);
  float* vb      = alloc(SME);
  float* attn    = alloc(SME);
  float* enc_out = alloc(SME);
  float* xd      = alloc(SME);
  float* scores  = alloc((size_t)M * FFd);  // union: scores (B,H,T,S) / FFN inner
  float* ff1     = scores;
  float* qp      = alloc((size_t)M * PROJ);
  float* kp      = alloc((size_t)M * PROJ);

  auto gemm = [&](bool transb, const float* A, const float* Bm, const float* bias,
                  const float* Cadd, float* C, int Mm, int Nn, int Kk,
                  long lda, long ldb, long ldc, int Z, int Z2,
                  long Az1, long Az2, long Bz1, long Bz2, long Cz1, long Cz2,
                  float alpha, int relu) {
    dim3 g((Nn + 63) / 64, (Mm + 63) / 64, Z);
    if (transb)
      gemm_bf16_wmma<true><<<g, 128, 0, stream>>>(A, Bm, bias, Cadd, C, Mm, Nn, Kk,
          lda, ldb, ldc, Z2, Az1, Az2, Bz1, Bz2, Cz1, Cz2, alpha, relu);
    else
      gemm_bf16_wmma<false><<<g, 128, 0, stream>>>(A, Bm, bias, Cadd, C, Mm, Nn, Kk,
          lda, ldb, ldc, Z2, Az1, Az2, Bz1, Bz2, Cz1, Cz2, alpha, relu);
  };

  const long TE  = (long)T * E;
  const long THD = (long)T * HD;
  const long EHD = (long)E * HD;
  const long ST  = (long)T * T;
  const float scale = sqrtf((float)HD);  // faithful: reference MULTIPLIES by sqrt(HD)

  auto mha = [&](const float* lnq, const float* lnkv_src,
                 const float* Wq, const float* Wk, const float* Wv, float* state) {
    gemm(false, lnq,      Wq, nullptr, nullptr, qb, T, HD, E, E, HD, HD,
         Bn * Hh, Hh, TE, 0, 0, EHD, (long)Hh * THD, THD, 1.f, 0);
    gemm(false, lnkv_src, Wk, nullptr, nullptr, kb, T, HD, E, E, HD, HD,
         Bn * Hh, Hh, TE, 0, 0, EHD, (long)Hh * THD, THD, 1.f, 0);
    gemm(false, lnkv_src, Wv, nullptr, nullptr, vb, T, HD, E, E, HD, HD,
         Bn * Hh, Hh, TE, 0, 0, EHD, (long)Hh * THD, THD, 1.f, 0);
    gemm(true, qb, kb, nullptr, nullptr, scores, T, T, HD, HD, HD, T,
         Bn * Hh, Hh, (long)Hh * THD, THD, (long)Hh * THD, THD,
         (long)Hh * ST, ST, scale, 0);
    softmax_kernel<<<Bn * Hh * T, 256, 0, stream>>>(scores, T);
    gemm(false, scores, vb, nullptr, nullptr, attn, T, HD, T, T, HD, E,
         Bn * Hh, Hh, (long)Hh * ST, ST, (long)Hh * THD, THD, TE, (long)HD, 1.f, 0);
    add_kernel<<<4096, 256, 0, stream>>>(state, attn, (long)M * E);
  };

  // ================= Encoder =================
  embed_src_kernel<<<dim3(T, Bn), 256, 0, stream>>>(Fp(119), src_idx, xe, T, E);
  for (int l = 0; l < 4; ++l) {
    const int p = 64 + 11 * l;
    layernorm_kernel<<<M, 256, 0, stream>>>(xe, Fp(p + 5), Fp(p + 4), lnb, E);
    mha(lnb, lnb, Fp(p + 9), Fp(p + 8), Fp(p + 10), xe);
    layernorm_kernel<<<M, 256, 0, stream>>>(xe, Fp(p + 7), Fp(p + 6), lnb, E);
    gemm(false, lnb, Fp(p + 0), Fp(p + 2), nullptr, ff1, M, FFd, E, E, FFd, FFd,
         1, 1, 0, 0, 0, 0, 0, 0, 1.f, 1);
    gemm(false, ff1, Fp(p + 1), Fp(p + 3), xe, xe, M, E, FFd, FFd, E, E,
         1, 1, 0, 0, 0, 0, 0, 0, 1.f, 0);
  }
  layernorm_kernel<<<M, 256, 0, stream>>>(xe, Fp(109), Fp(108), enc_out, E);

  // ================= Decoder =================
  embed_dec_kernel<<<dim3(T, Bn), 256, 0, stream>>>(Fp(122), Fp(110), Fp(123),
      tgt_idx, field_idx, type_idx, xd, T, E3);
  for (int l = 0; l < 4; ++l) {
    const int p = 16 * l;
    layernorm_kernel<<<M, 256, 0, stream>>>(xd, Fp(p + 8), Fp(p + 7), lnb, E);
    mha(lnb, lnb, Fp(p + 14), Fp(p + 13), Fp(p + 15), xd);
    layernorm_kernel<<<M, 256, 0, stream>>>(xd,      Fp(p + 10), Fp(p + 9), lnb,  E);
    layernorm_kernel<<<M, 256, 0, stream>>>(enc_out, Fp(p + 10), Fp(p + 9), lnkv, E);
    mha(lnb, lnkv, Fp(p + 1), Fp(p + 0), Fp(p + 2), xd);
    layernorm_kernel<<<M, 256, 0, stream>>>(xd, Fp(p + 12), Fp(p + 11), lnb, E);
    gemm(false, lnb, Fp(p + 3), Fp(p + 5), nullptr, ff1, M, FFd, E, E, FFd, FFd,
         1, 1, 0, 0, 0, 0, 0, 0, 1.f, 1);
    gemm(false, ff1, Fp(p + 4), Fp(p + 6), xd, xd, M, E, FFd, FFd, E, E,
         1, 1, 0, 0, 0, 0, 0, 0, 1.f, 0);
  }

  // ================= Heads =================
  float* out = (float*)d_out;
  float* out_gc   = out;            //  8*512*2
  float* out_ptr  = out + 8192;     //  8*512*512
  float* out_tgt  = out + 2105344;  //  8*512*512
  float* out_prim = out + 4202496;  //  8*512*10000

  gen_copy_kernel<<<M, 256, 0, stream>>>(xd, Fp(111), Fp(112), out_gc, E);

  gemm(false, xd,      Fp(115),                  Fp(117), nullptr, qp, M, PROJ, E,
       E, PROJ, PROJ, 1, 1, 0, 0, 0, 0, 0, 0, 1.f, 0);
  gemm(false, enc_out, Fp(115) + (long)E * PROJ, nullptr, nullptr, kp, M, PROJ, E,
       E, PROJ, PROJ, 1, 1, 0, 0, 0, 0, 0, 0, 1.f, 0);
  ptr_kernel<<<dim3(T, Bn), 256, 0, stream>>>(qp, kp, Fp(116), Fp(118), out_ptr,
                                              T, T, PROJ);

  gemm(false, xd, Fp(120), Fp(121), nullptr, out_tgt, M, TGTV, E, E, TGTV, TGTV,
       1, 1, 0, 0, 0, 0, 0, 0, 1.f, 0);
  gemm(false, xd, Fp(113), Fp(114), nullptr, out_prim, M, PRIMV, E, E, PRIMV, PRIMV,
       1, 1, 0, 0, 0, 0, 0, 0, 1.f, 0);
}
